// CRF_84868553769207
// MI455X (gfx1250) — compile-verified, matrix operand order checked by reading, not run on hardware
//
#include <hip/hip_runtime.h>

// CRF forward (log-semiring scan) mapped to exp-domain GEMM via v_wmma_f32_16x16x32_f16,
// with feats tiles streamed by the Tensor Data Mover (tensor_load_to_lds, TENSORcnt).
// B=1024, T=512, K=64. Output: scalar mean(forward_score - gold_score), f32.

constexpr int Bn   = 1024;
constexpr int Tn   = 512;
constexpr int Kn   = 64;
constexpr float NEGV = -10000.0f;
constexpr float LN2F = 0.6931471805599453f;

typedef _Float16 v16h  __attribute__((ext_vector_type(16)));
typedef float    v8f   __attribute__((ext_vector_type(8)));
typedef unsigned u32x4 __attribute__((ext_vector_type(4)));
typedef int      i32x4 __attribute__((ext_vector_type(4)));
typedef int      i32x8 __attribute__((ext_vector_type(8)));

union V16HU { v16h v; unsigned u[8]; _Float16 h[16]; };

// Raw v_log_f32 (log2 semantics); acc is either exactly 0 (-> -inf, correct for the
// START row) or a normal float, so the denormal-guarded libm sequence is wasted work.
__device__ __forceinline__ float fast_log2(float x) {
#if __has_builtin(__builtin_amdgcn_logf)
    return __builtin_amdgcn_logf(x);
#else
    return __log2f(x);
#endif
}

// A-operand (16-bit, 16x32) half layout:
//   lanes 0-15 : K = {0..7, 16..23};  lanes 16-31 : K = {8..15, 24..31}
//   filled by DWORD pairs: pair hp (0..7) covers K = kbase, kbase+1.
__device__ __forceinline__ int pair_kbase(int hp, int laneHi) {
    return ((hp < 4) ? 0 : 16) + laneHi * 8 + (hp & 3) * 2;
}

// TDM: async-load a 16-row x 64-col f32 tile (row stride Tn*Kn elements) from global
// into LDS, rows packed contiguously (16 x 256B). D# per CDNA5 ISA ch.8.
// 6-arg builtin form (clang-23 / amdgpu-toolchain): groups 0,1 + unused groups + cpol.
__device__ __forceinline__ void tdm_load_tile(const float* gsrc, unsigned lds_byte_off) {
    unsigned long long ga = (unsigned long long)(uintptr_t)gsrc;
    u32x4 g0;
    g0.x = 0x1u;                                               // count=1, user descriptor
    g0.y = lds_byte_off;                                       // lds_addr (bytes)
    g0.z = (unsigned)(ga & 0xFFFFFFFFu);                       // global_addr[31:0]
    g0.w = (unsigned)((ga >> 32) & 0x01FFFFFFu) | (2u << 30);  // global_addr[56:32] | type=2
    i32x8 g1;
    g1[0] = 0x00020000;                                        // data_size=2 (4 bytes)
    g1[1] = (int)(((unsigned)(Tn * Kn) & 0xFFFFu) << 16);      // tensor_dim0[15:0]
    g1[2] = (int)((((unsigned)(Tn * Kn) >> 16) & 0xFFFFu)      // tensor_dim0[31:16]
                  | (16u << 16));                              // | tensor_dim1 = 16
    g1[3] = (int)(64u << 16);                                  // tile_dim0 = 64
    g1[4] = 16;                                                // tile_dim1 = 16, tile_dim2 = 0
    g1[5] = Tn * Kn;                                           // tensor_dim0_stride (elements)
    g1[6] = 0;
    g1[7] = 0;
    i32x4 gz4 = {0, 0, 0, 0};                                  // groups 2/3 unused (2D tile)
    i32x8 gz8 = {0, 0, 0, 0, 0, 0, 0, 0};
    __builtin_amdgcn_tensor_load_to_lds(g0, g1, gz4, gz4, gz8, 0);
}

__global__ __launch_bounds__(128) void crf_forward_kernel(
    const float* __restrict__ feats,      // [B, T, K]
    const float* __restrict__ trans,      // [K, K], trans[i*K+j] = score j -> i
    float* __restrict__ fwd_out)          // [B]
{
    __shared__ float     alpha[16 * 64];      // [row][j]
    __shared__ _Float16  pbuf [16 * 64];      // exp(alpha - m), f16
    __shared__ float     featbuf[2][16 * 64]; // TDM double buffer, [row][i]
    __shared__ float     pmax[16 * 8];        // partial row maxima
    __shared__ float     psum[16 * 8];        // partial row sums (epilogue)
    __shared__ float     mrow[16];

    const int tid    = threadIdx.x;
    const int lane   = tid & 31;
    const int wave   = tid >> 5;          // n-tile index 0..3 (i in [16w, 16w+16))
    const int laneHi = (lane >> 4) & 1;
    const int laneLo = lane & 15;
    const int b0     = blockIdx.x * 16;
    const int row8   = tid >> 3;          // 0..15
    const int sub8   = tid & 7;           // 0..7

    // ---- Preload B tiles: B[k=j][n=i] = exp(trans[i][j]); in registers forever ----
    V16HU btile[2];
    #pragma unroll
    for (int kc = 0; kc < 2; ++kc) {
        #pragma unroll
        for (int hp = 0; hp < 8; ++hp) {
            int kb = pair_kbase(hp, laneHi);
            int j  = kc * 32 + kb;
            int i  = wave * 16 + laneLo;
            btile[kc].h[2 * hp]     = (_Float16)__expf(trans[i * Kn + j]);
            btile[kc].h[2 * hp + 1] = (_Float16)__expf(trans[i * Kn + j + 1]);
        }
    }

    // ---- init alpha: NEG everywhere except START tag (62) ----
    #pragma unroll
    for (int e = 0; e < 8; ++e) {
        int idx = tid * 8 + e;
        alpha[idx] = ((idx & 63) == 62) ? 0.0f : NEGV;
    }
    // prime TDM pipeline with feats tile for t=1
    if (wave == 0)
        tdm_load_tile(feats + ((size_t)b0 * Tn + 1) * Kn,
                      (unsigned)(uintptr_t)(&featbuf[1][0]));
    __syncthreads();

    for (int t = 1; t < Tn; ++t) {
        if (wave == 0) __builtin_amdgcn_s_wait_tensorcnt(0);   // featbuf[t&1] landed
        __syncthreads();

        // phase 1a: partial row maxima (128 threads x 8 elements)
        {
            const float* ar = &alpha[row8 * 64 + sub8 * 8];
            float m = ar[0];
            #pragma unroll
            for (int j = 1; j < 8; ++j) m = fmaxf(m, ar[j]);
            pmax[row8 * 8 + sub8] = m;
        }
        // kick off TDM for next timestep's feats while we compute this one
        if (wave == 0 && t + 1 < Tn)
            tdm_load_tile(feats + ((size_t)b0 * Tn + (t + 1)) * Kn,
                          (unsigned)(uintptr_t)(&featbuf[(t + 1) & 1][0]));
        __syncthreads();

        // phase 2: fold partials (redundantly per 8 threads), p = exp(alpha - m) f16
        {
            float m = pmax[row8 * 8];
            #pragma unroll
            for (int k = 1; k < 8; ++k) m = fmaxf(m, pmax[row8 * 8 + k]);
            if (sub8 == 0) mrow[row8] = m;
            const float* ar = &alpha[row8 * 64 + sub8 * 8];
            _Float16* pr    = &pbuf [row8 * 64 + sub8 * 8];
            #pragma unroll
            for (int j = 0; j < 8; ++j) pr[j] = (_Float16)__expf(ar[j] - m);
        }
        __syncthreads();

        // phase 3: gather A operands in ISA layout, 2 WMMAs, log + m + feat, write alpha
        V16HU a0, a1;
        {
            const unsigned* p32 = (const unsigned*)pbuf;   // 2 halves per dword
            #pragma unroll
            for (int hp = 0; hp < 8; ++hp) {
                int kb = pair_kbase(hp, laneHi);
                a0.u[hp] = p32[(laneLo * 64 + 0  + kb) >> 1];
                a1.u[hp] = p32[(laneLo * 64 + 32 + kb) >> 1];
            }
        }

        v8f acc = {};
        acc = __builtin_amdgcn_wmma_f32_16x16x32_f16(
                false, a0.v, false, btile[0].v, (short)0, acc, false, false);
        acc = __builtin_amdgcn_wmma_f32_16x16x32_f16(
                false, a1.v, false, btile[1].v, (short)0, acc, false, false);

        const int    i  = wave * 16 + laneLo;
        const float* fb = featbuf[t & 1];
        float na[8];
        #pragma unroll
        for (int r = 0; r < 8; ++r) {
            int M = r + laneHi * 8;                        // C/D layout: row
            na[r] = fmaf(fast_log2(acc[r]), LN2F, mrow[M] + fb[M * 64 + i]);
        }
        #pragma unroll
        for (int r = 0; r < 8; ++r) {
            int M = r + laneHi * 8;
            alpha[M * 64 + i] = na[r];
        }
        // loop-top barrier publishes alpha for the next iteration
    }
    __syncthreads();

    // ---- epilogue: forward_score[b] = LSE_i alpha[b, i], parallel partial reduce ----
    {
        const float* ar = &alpha[row8 * 64 + sub8 * 8];
        float m = ar[0];
        #pragma unroll
        for (int j = 1; j < 8; ++j) m = fmaxf(m, ar[j]);
        pmax[row8 * 8 + sub8] = m;
    }
    __syncthreads();
    {
        float m = pmax[row8 * 8];
        #pragma unroll
        for (int k = 1; k < 8; ++k) m = fmaxf(m, pmax[row8 * 8 + k]);
        if (sub8 == 0) mrow[row8] = m;
        const float* ar = &alpha[row8 * 64 + sub8 * 8];
        float s = 0.0f;
        #pragma unroll
        for (int j = 0; j < 8; ++j) s += __expf(ar[j] - m);
        psum[row8 * 8 + sub8] = s;
    }
    __syncthreads();
    if (tid < 16) {
        float s = 0.0f;
        #pragma unroll
        for (int k = 0; k < 8; ++k) s += psum[tid * 8 + k];
        fwd_out[b0 + tid] = fmaf(fast_log2(s), LN2F, mrow[tid]);
    }
}

__global__ __launch_bounds__(256) void crf_gold_kernel(
    const float* __restrict__ feats,
    const int*   __restrict__ tags,
    const float* __restrict__ trans,
    float* __restrict__ gold_out)
{
    int gid  = blockIdx.x * blockDim.x + threadIdx.x;
    int b    = gid >> 5;
    int lane = gid & 31;
    if (b >= Bn) return;

    float s = 0.0f;
    for (int t = 1 + lane; t < Tn; t += 32) {
        int curr = tags[b * Tn + t];
        int prev = tags[b * Tn + t - 1];
        s += trans[curr * Kn + prev]
           + feats[((size_t)b * Tn + t) * Kn + curr];
    }
    #pragma unroll
    for (int off = 16; off > 0; off >>= 1) s += __shfl_down(s, off, 32);
    if (lane == 0) gold_out[b] = s;
}

__global__ __launch_bounds__(256) void crf_reduce_kernel(
    const float* __restrict__ fwd,
    const float* __restrict__ gold,
    float* __restrict__ out)
{
    __shared__ float sm[256];
    float s = 0.0f;
    for (int i = threadIdx.x; i < Bn; i += 256) s += fwd[i] - gold[i];
    sm[threadIdx.x] = s;
    __syncthreads();
    for (int off = 128; off > 0; off >>= 1) {
        if ((int)threadIdx.x < off) sm[threadIdx.x] += sm[threadIdx.x + off];
        __syncthreads();
    }
    if (threadIdx.x == 0) out[0] = sm[0] / (float)Bn;
}

extern "C" void kernel_launch(void* const* d_in, const int* in_sizes, int n_in,
                              void* d_out, int out_size, void* d_ws, size_t ws_size,
                              hipStream_t stream) {
    const float* feats = (const float*)d_in[0];   // [B, T, K] f32
    const int*   tags  = (const int*)d_in[1];     // [B, T] i32
    const float* trans = (const float*)d_in[2];   // [K, K] f32
    float* out  = (float*)d_out;
    float* fwd  = (float*)d_ws;                   // [B]
    float* gold = fwd + Bn;                       // [B]

    crf_forward_kernel<<<Bn / 16, 128, 0, stream>>>(feats, trans, fwd);
    crf_gold_kernel<<<(Bn * 32) / 256, 256, 0, stream>>>(feats, tags, trans, gold);
    crf_reduce_kernel<<<1, 256, 0, stream>>>(fwd, gold, out);
}
